// DiM_13726715478168
// MI455X (gfx1250) — matrix operand order repaired and checked by reading, not compiled, not run on hardware
//
#include <hip/hip_runtime.h>
#include <hip/hip_bf16.h>
#include <math.h>

// ---------------- model dims ----------------
#define NB   8      // batch
#define CIN  4
#define IMG  28
#define PP   2      // patch
#define DD   512
#define DEPTH 8
#define RR   32
#define SS   16
#define EE   1024   // 2*DD
#define GG   14     // IMG/PP
#define LL   196    // GG*GG
#define COUT 8
#define MROWS (NB*LL)   // 1568

typedef __bf16 bf16_t;
typedef __attribute__((ext_vector_type(16))) __bf16 v16bf;
typedef __attribute__((ext_vector_type(8)))  float  v8f;

// ---------------- helpers ----------------
__device__ __forceinline__ bf16_t to_bf16(float f) {
    // round-to-nearest-even f32 -> bf16
    union { float f; unsigned u; } x; x.f = f;
    unsigned r = (x.u + 0x7FFFu + ((x.u >> 16) & 1u)) >> 16;
    unsigned short hs = (unsigned short)r;
    bf16_t b;
    __builtin_memcpy(&b, &hs, 2);
    return b;
}
__device__ __forceinline__ float siluf(float x)     { return x / (1.f + __expf(-x)); }
__device__ __forceinline__ float softplusf(float x) { return (x > 20.f) ? x : log1pf(__expf(x)); }

#define LOG1E4 9.2103403719761836f   // ln(10000)

// ---------------- spiral order (1 thread) ----------------
__global__ void k_make_order(int* __restrict__ ord, int* __restrict__ inv) {
    int t = 0, b = GG - 1, l = 0, r = GG - 1, c = 0;
    while (t <= b && l <= r) {
        for (int j = l; j <= r; ++j) ord[c++] = t * GG + j;
        ++t; if (t > b) break;
        for (int i = t; i <= b; ++i) ord[c++] = i * GG + r;
        --r; if (l > r) break;
        for (int j = r; j >= l; --j) ord[c++] = b * GG + j;
        --b; if (t > b) break;
        for (int i = b; i >= t; --i) ord[c++] = i * GG + l;
        ++l;
    }
    for (int i = 0; i < LL; ++i) inv[ord[i]] = i;
}

// ---------------- patch embed + pos embed + bias ----------------
__global__ void k_patch_embed(const float* __restrict__ x, const float* __restrict__ pw,
                              const float* __restrict__ pb, float* __restrict__ h) {
    int idx = blockIdx.x * blockDim.x + threadIdx.x;
    if (idx >= NB * LL * DD) return;
    int d = idx % DD;
    int l = (idx / DD) % LL;
    int n = idx / (DD * LL);
    int gh = l / GG, gw = l % GG;
    float acc = pb[d];
#pragma unroll
    for (int ci = 0; ci < CIN; ++ci)
#pragma unroll
        for (int p = 0; p < PP; ++p)
#pragma unroll
            for (int q = 0; q < PP; ++q)
                acc += x[((n * CIN + ci) * IMG + gh * PP + p) * IMG + gw * PP + q] *
                       pw[((d * CIN + ci) * PP + p) * PP + q];
    // 2D sincos pos-embed: [sin(px*w), cos(px*w), sin(py*w), cos(py*w)] blocks of 128
    int grp = d >> 7, i = d & 127;
    float om  = __expf(-(float)i * (LOG1E4 / 128.f));
    float pos = (grp < 2) ? (float)(l % GG) : (float)(l / GG);
    float a   = pos * om;
    float pe  = (grp & 1) ? __cosf(a) : __sinf(a);
    h[idx] = acc + pe;
}

// ---------------- timestep fourier features ----------------
__global__ void k_time_base(const float* __restrict__ t, float* __restrict__ out) {
    int idx = blockIdx.x * blockDim.x + threadIdx.x;
    if (idx >= NB * 256) return;
    int n = idx / 256, k = idx % 256;
    int kk = k & 127;
    float f = __expf(-(float)kk * (LOG1E4 / 128.f));
    float a = t[n] * f;
    out[idx] = (k < 128) ? __cosf(a) : __sinf(a);
}

// ---------------- small (M=8) GEMM, one thread per output ----------------
__global__ void k_small_gemm(const float* __restrict__ A, const float* __restrict__ B,
                             const float* __restrict__ bias, float* __restrict__ C,
                             int M, int N, int K, int act) {
    int idx = blockIdx.x * blockDim.x + threadIdx.x;
    if (idx >= M * N) return;
    int m = idx / N, j = idx % N;
    float acc = bias ? bias[j] : 0.f;
    const float* a = A + (size_t)m * K;
    for (int k = 0; k < K; ++k) acc += a[k] * B[(size_t)k * N + j];
    if (act == 1) acc = siluf(acc);
    C[idx] = acc;
}

__global__ void k_build_cact(const float* __restrict__ temb, const float* __restrict__ y,
                             float* __restrict__ cact) {
    int idx = blockIdx.x * blockDim.x + threadIdx.x;
    if (idx >= NB * 2 * DD) return;
    int n = idx >> 10, j = idx & 1023;
    float v = (j < DD) ? temb[n * DD + j] : y[n * DD + (j - DD)];
    cact[idx] = siluf(v);
}

// ---------------- f32 -> bf16 (vectorized x4) ----------------
__global__ void k_cvt_bf16(const float* __restrict__ in, bf16_t* __restrict__ out, int n4) {
    int i = blockIdx.x * blockDim.x + threadIdx.x;
    if (i >= n4) return;
    float4 v = ((const float4*)in)[i];
    union { bf16_t b[4]; uint2 u; } o;
    o.b[0] = to_bf16(v.x); o.b[1] = to_bf16(v.y);
    o.b[2] = to_bf16(v.z); o.b[3] = to_bf16(v.w);
    ((uint2*)out)[i] = o.u;
}

// ---------------- transpose + convert: B (KxN f32) -> Bt (NxK bf16) ----------------
__global__ void k_tconv_bf16(const float* __restrict__ B, bf16_t* __restrict__ Bt,
                             int K, int N) {
    int idx = blockIdx.x * blockDim.x + threadIdx.x;
    if (idx >= K * N) return;
    int k = idx / N, n = idx % N;
    Bt[(size_t)n * K + k] = to_bf16(B[idx]);
}

// ---------------- LayerNorm + modulate (writes f32 OR bf16) ----------------
__global__ void k_ln_mod(const float* __restrict__ h, const float* __restrict__ sg,
                         int sgStride, int shiftOff, int scaleOff,
                         float* __restrict__ outf, bf16_t* __restrict__ outbf) {
    __shared__ float s1[128], s2[128];
    int row = blockIdx.x;            // n*LL + l
    int n   = row / LL;
    int t   = threadIdx.x;           // 128 threads, 4 elems each
    const float* hp = h + (size_t)row * DD;
    float v[4], a0 = 0.f, a1 = 0.f;
#pragma unroll
    for (int i = 0; i < 4; ++i) { float x = hp[t * 4 + i]; v[i] = x; a0 += x; a1 += x * x; }
    s1[t] = a0; s2[t] = a1;
    __syncthreads();
    for (int off = 64; off > 0; off >>= 1) {
        if (t < off) { s1[t] += s1[t + off]; s2[t] += s2[t + off]; }
        __syncthreads();
    }
    float mu   = s1[0] * (1.f / DD);
    float var  = s2[0] * (1.f / DD) - mu * mu;
    float rstd = rsqrtf(var + 1e-6f);
    const float* sh = sg + (size_t)n * sgStride;
#pragma unroll
    for (int i = 0; i < 4; ++i) {
        int d = t * 4 + i;
        float xn = (v[i] - mu) * rstd;
        float o  = xn * (1.f + sh[scaleOff + d]) + sh[shiftOff + d];
        if (outbf) outbf[(size_t)row * DD + d] = to_bf16(o);
        else       outf [(size_t)row * DD + d] = o;
    }
}

// ============ blocked WMMA bf16 GEMM: C = A(MxK,bf16) * Bt(NxK,bf16)^T + bias =========
// One wave computes a 32x64 tile: 2 A-frags x 4 B-frags -> 8 WMMAs per K-step of 32.
// A row-major bf16; Bt is B transposed (row n holds K contiguous) -> all frags are b128 loads.
union frag_u { uint4 q[2]; v16bf v; };

__global__ __launch_bounds__(32) void k_wmma_gemm32x64(
        const bf16_t* __restrict__ A, const bf16_t* __restrict__ Bt,
        const float* __restrict__ bias, float* __restrict__ C,
        int M, int N, int K, int lda, int ldbt, int ldc, int act) {
    int ntn  = N >> 6;
    int tm   = (int)blockIdx.x / ntn;
    int tn   = (int)blockIdx.x % ntn;
    int lane = threadIdx.x;          // 0..31
    int r    = lane & 15;
    int hi   = lane >> 4;

    // A frag (16-bit A layout, ISA 7.12.2): lane holds K {k0+hi*8+0..7, k0+hi*8+16..23}
    const bf16_t* a0 = A + (size_t)(tm * 32 + r) * lda + hi * 8;
    const bf16_t* a1 = a0 + (size_t)16 * lda;
    // B frag: lane holds col (tn*64 + j*16 + r), K {k0+hi*8+0..7, k0+hi*8+16..23} in Bt
    const bf16_t* b0 = Bt + (size_t)(tn * 64 + r) * ldbt + hi * 8;

    v8f acc[2][4] = {};
    for (int k0 = 0; k0 < K; k0 += 32) {
        frag_u af[2], bf[4];
        af[0].q[0] = *(const uint4*)(a0 + k0);
        af[0].q[1] = *(const uint4*)(a0 + k0 + 16);
        af[1].q[0] = *(const uint4*)(a1 + k0);
        af[1].q[1] = *(const uint4*)(a1 + k0 + 16);
#pragma unroll
        for (int j = 0; j < 4; ++j) {
            const bf16_t* bp = b0 + (size_t)j * 16 * ldbt + k0;
            bf[j].q[0] = *(const uint4*)(bp);
            bf[j].q[1] = *(const uint4*)(bp + 16);
        }
        __builtin_prefetch((const void*)(a0 + k0 + 32), 0, 0);
        __builtin_prefetch((const void*)(b0 + k0 + 32), 0, 0);
#pragma unroll
        for (int i = 0; i < 2; ++i)
#pragma unroll
            for (int j = 0; j < 4; ++j)
                acc[i][j] = __builtin_amdgcn_wmma_f32_16x16x32_bf16(
                    false, af[i].v, false, bf[j].v, (short)0, acc[i][j], false, false);
    }
    // C/D layout: col = lane&15 within each 16-col subtile, row = v + 8*(lane>>4)
    float bv[4];
#pragma unroll
    for (int j = 0; j < 4; ++j) bv[j] = bias ? bias[tn * 64 + j * 16 + r] : 0.f;
#pragma unroll
    for (int i = 0; i < 2; ++i) {
#pragma unroll
        for (int v = 0; v < 8; ++v) {
            int m = tm * 32 + i * 16 + v + 8 * hi;
            float* cp = C + (size_t)m * ldc + tn * 64 + r;
#pragma unroll
            for (int j = 0; j < 4; ++j) {
                float x = acc[i][j][v] + bv[j];
                if (act == 1) x = softplusf(x);
                cp[j * 16] = x;
            }
        }
    }
}

// ---------------- single-tile WMMA GEMM on f32 inputs (small-N head) ----------------
__global__ __launch_bounds__(32) void k_wmma_gemm16(
        const float* __restrict__ A, const float* __restrict__ B,
        const float* __restrict__ bias, float* __restrict__ C,
        int M, int N, int K, int lda, int ldb, int ldc, int act) {
    int ntn  = N >> 4;
    int tm   = (int)blockIdx.x / ntn;
    int tn   = (int)blockIdx.x % ntn;
    int lane = threadIdx.x;
    int r    = lane & 15;
    int hi   = lane >> 4;
    const float* Arow = A + (size_t)(tm * 16 + r) * lda;
    const float* Bcol = B + (tn * 16 + r);
    v8f acc = {};
    for (int k0 = 0; k0 < K; k0 += 32) {
        v16bf a, b;
        const float4* ap0 = (const float4*)(Arow + k0 + hi * 8);
        const float4* ap1 = (const float4*)(Arow + k0 + hi * 8 + 16);
        float4 x0 = ap0[0], x1 = ap0[1], x2 = ap1[0], x3 = ap1[1];
        a[0]  = to_bf16(x0.x); a[1]  = to_bf16(x0.y); a[2]  = to_bf16(x0.z); a[3]  = to_bf16(x0.w);
        a[4]  = to_bf16(x1.x); a[5]  = to_bf16(x1.y); a[6]  = to_bf16(x1.z); a[7]  = to_bf16(x1.w);
        a[8]  = to_bf16(x2.x); a[9]  = to_bf16(x2.y); a[10] = to_bf16(x2.z); a[11] = to_bf16(x2.w);
        a[12] = to_bf16(x3.x); a[13] = to_bf16(x3.y); a[14] = to_bf16(x3.z); a[15] = to_bf16(x3.w);
#pragma unroll
        for (int j = 0; j < 16; ++j)
            b[j] = to_bf16(Bcol[(size_t)(k0 + hi * 8 + (j & 7) + ((j >> 3) << 4)) * ldb]);
        acc = __builtin_amdgcn_wmma_f32_16x16x32_bf16(false, a, false, b,
                                                      (short)0, acc, false, false);
    }
    int c = tn * 16 + r;
    float bvv = bias ? bias[c] : 0.f;
#pragma unroll
    for (int v = 0; v < 8; ++v) {
        int m = tm * 16 + v + 8 * hi;
        float x = acc[v] + bvv;
        if (act == 1) x = softplusf(x);
        C[(size_t)m * ldc + c] = x;
    }
}

// ---------------- z = uz[:,:,E:] + zc*w ; silu ----------------
__global__ void k_zsilu(const float* __restrict__ uz, const float* __restrict__ zc,
                        const float* __restrict__ w, float* __restrict__ siluz) {
    int idx = blockIdx.x * blockDim.x + threadIdx.x;
    if (idx >= MROWS * EE) return;
    int row = idx >> 10;
    float z = uz[(size_t)row * (2 * EE) + EE + (idx & 1023)] + zc[idx] * w[row];
    siluz[idx] = siluf(z);
}

// ---------------- u_s = u[:, order] (writes f32 for scan + bf16 for GEMM) ----------------
__global__ void k_gather(const float* __restrict__ uz, const int* __restrict__ order,
                         float* __restrict__ us, bf16_t* __restrict__ usbf) {
    int idx = blockIdx.x * blockDim.x + threadIdx.x;
    if (idx >= MROWS * EE) return;
    int e  = idx & 1023;
    int ls = (idx >> 10) % LL;
    int n  = idx / (LL * EE);
    float v = uz[((size_t)(n * LL + order[ls])) * (2 * EE) + e];
    us[idx]   = v;
    usbf[idx] = to_bf16(v);
}

// ---------------- selective scan (one thread per (n,e), 16 states) ----------------
__global__ void k_scan(const float* __restrict__ us, const float* __restrict__ dtb,
                       const float* __restrict__ dbc, const float* __restrict__ Alog,
                       float* __restrict__ yout, int dir) {
    int tid = blockIdx.x * blockDim.x + threadIdx.x;
    if (tid >= NB * EE) return;
    int n = tid >> 10, e = tid & 1023;
    float As[SS], st[SS];
#pragma unroll
    for (int s = 0; s < SS; ++s) { As[s] = -__expf(Alog[e * SS + s]); st[s] = 0.f; }
    for (int step = 0; step < LL; ++step) {
        int l = dir ? (LL - 1 - step) : step;
        size_t base = (size_t)n * LL + l;
        float dt = dtb[base * EE + e];
        float u  = us [base * EE + e];
        float du = dt * u;
        const float* bc = dbc + base * (RR + 2 * SS);
        float y = 0.f;
#pragma unroll
        for (int s = 0; s < SS; ++s) {
            float stv = __expf(dt * As[s]) * st[s] + du * bc[RR + s];
            st[s] = stv;
            y += stv * bc[RR + SS + s];
        }
        yout[base * EE + e] = y;
    }
}

// ---------------- ys reorder + D skip + gate by silu(z) -> bf16 GEMM input ----------------
__global__ void k_combine(const float* __restrict__ yf, const float* __restrict__ yr,
                          const float* __restrict__ us, const float* __restrict__ Dsk,
                          const float* __restrict__ siluz, const int* __restrict__ inv,
                          bf16_t* __restrict__ aoutbf) {
    int idx = blockIdx.x * blockDim.x + threadIdx.x;
    if (idx >= MROWS * EE) return;
    int e = idx & 1023;
    int l = (idx >> 10) % LL;
    int n = idx / (LL * EE);
    size_t s = ((size_t)(n * LL + inv[l])) * EE + e;
    aoutbf[idx] = to_bf16((yf[s] + yr[s] + us[s] * Dsk[e]) * siluz[idx]);
}

// ---------------- residual: h += gate * out ----------------
__global__ void k_residual(float* __restrict__ h, const float* __restrict__ sg,
                           const float* __restrict__ outb) {
    int idx = blockIdx.x * blockDim.x + threadIdx.x;
    if (idx >= MROWS * DD) return;
    int n = idx / (LL * DD);
    int d = idx & 511;
    h[idx] += sg[n * 3 * DD + 2 * DD + d] * outb[idx];
}

// ---------------- unpatchify ----------------
__global__ void k_unpatchify(const float* __restrict__ proj, float* __restrict__ out) {
    int idx = blockIdx.x * blockDim.x + threadIdx.x;
    if (idx >= NB * COUT * IMG * IMG) return;
    int wq = idx % IMG;
    int hp = (idx / IMG) % IMG;
    int c  = (idx / (IMG * IMG)) % COUT;
    int n  = idx / (COUT * IMG * IMG);
    int gh = hp / PP, p = hp % PP;
    int gw = wq / PP, q = wq % PP;
    out[idx] = proj[((size_t)(n * LL + gh * GG + gw)) * (PP * PP * COUT) + (p * PP + q) * COUT + c];
}

// ---------------- workspace layout ----------------
static constexpr size_t W_ORD  = 0;     // int[196]
static constexpr size_t W_INV  = 196;   // int[196]
static constexpr size_t F_H     = 512;
static constexpr size_t F_TIN   = F_H    + (size_t)MROWS * DD;
static constexpr size_t F_T1    = F_TIN  + (size_t)NB * 256;
static constexpr size_t F_TEMB  = F_T1   + (size_t)NB * DD;
static constexpr size_t F_CACT  = F_TEMB + (size_t)NB * DD;
static constexpr size_t F_SG    = F_CACT + (size_t)NB * 2 * DD;
static constexpr size_t F_FSG   = F_SG   + (size_t)NB * 3 * DD;
static constexpr size_t F_HM    = F_FSG  + (size_t)NB * 2 * DD;
static constexpr size_t F_UZ    = F_HM   + (size_t)MROWS * DD;
static constexpr size_t F_ZC    = F_UZ   + (size_t)MROWS * 2 * EE;
static constexpr size_t F_SILUZ = F_ZC   + (size_t)MROWS * EE;
static constexpr size_t F_US    = F_SILUZ+ (size_t)MROWS * EE;
static constexpr size_t F_DBC   = F_US   + (size_t)MROWS * EE;
static constexpr size_t F_DTB   = F_DBC  + (size_t)MROWS * (RR + 2 * SS);
static constexpr size_t F_YF    = F_DTB  + (size_t)MROWS * EE;
static constexpr size_t F_YR    = F_YF   + (size_t)MROWS * EE;
static constexpr size_t F_OUTB  = F_YR   + (size_t)MROWS * EE;
static constexpr size_t F_PROJ  = F_OUTB + (size_t)MROWS * DD;
static constexpr size_t F_END   = F_PROJ + (size_t)MROWS * PP * PP * COUT;
// bf16 region (element offsets into bf16_t* based at fw + F_END)
static constexpr size_t B_HM    = 0;
static constexpr size_t B_Y2    = B_HM    + (size_t)MROWS * DD;
static constexpr size_t B_US    = B_Y2    + (size_t)MROWS * DD;
static constexpr size_t B_DBC   = B_US    + (size_t)MROWS * EE;
static constexpr size_t B_AOUT  = B_DBC   + (size_t)MROWS * (RR + 2 * SS);
static constexpr size_t B_WTIN  = B_AOUT  + (size_t)MROWS * EE;                 // [8][2048*512]
static constexpr size_t B_WTCO  = B_WTIN  + (size_t)DEPTH * 2 * EE * DD;        // [8][1024*512]
static constexpr size_t B_WTXP  = B_WTCO  + (size_t)DEPTH * EE * DD;            // [8][64*1024]
static constexpr size_t B_WTDT  = B_WTXP  + (size_t)DEPTH * (RR + 2 * SS) * EE; // [8][1024*32]
static constexpr size_t B_WTOU  = B_WTDT  + (size_t)DEPTH * EE * RR;            // [8][512*1024]

#define CDIV(a, b) (((a) + (b) - 1) / (b))

extern "C" void kernel_launch(void* const* d_in, const int* in_sizes, int n_in,
                              void* d_out, int out_size, void* d_ws, size_t ws_size,
                              hipStream_t stream) {
    const float* x        = (const float*)d_in[0];
    const float* t        = (const float*)d_in[1];
    const float* y        = (const float*)d_in[2];
    const float* y2       = (const float*)d_in[3];
    const float* w        = (const float*)d_in[4];
    const float* patch_w  = (const float*)d_in[5];
    const float* patch_b  = (const float*)d_in[6];
    const float* t_w1     = (const float*)d_in[7];
    const float* t_b1     = (const float*)d_in[8];
    const float* t_w2     = (const float*)d_in[9];
    const float* t_b2     = (const float*)d_in[10];
    const float* b_ada_w  = (const float*)d_in[11];
    const float* b_ada_b  = (const float*)d_in[12];
    const float* b_in_w   = (const float*)d_in[13];
    const float* b_in_b   = (const float*)d_in[14];
    const float* b_cond_w = (const float*)d_in[15];
    const float* b_cond_b = (const float*)d_in[16];
    const float* b_xproj_w= (const float*)d_in[17];
    const float* b_dt_w   = (const float*)d_in[18];
    const float* b_dt_b   = (const float*)d_in[19];
    const float* b_Alog   = (const float*)d_in[20];
    const float* b_D      = (const float*)d_in[21];
    const float* b_out_w  = (const float*)d_in[22];
    const float* b_out_b  = (const float*)d_in[23];
    const float* f_ada_w  = (const float*)d_in[24];
    const float* f_ada_b  = (const float*)d_in[25];
    const float* f_lin_w  = (const float*)d_in[26];
    const float* f_lin_b  = (const float*)d_in[27];

    float*  fw   = (float*)d_ws;
    int*    iw   = (int*)d_ws;
    int*    ordp = iw + W_ORD;
    int*    invp = iw + W_INV;
    float*  h     = fw + F_H;
    float*  tin   = fw + F_TIN;
    float*  t1    = fw + F_T1;
    float*  temb  = fw + F_TEMB;
    float*  cact  = fw + F_CACT;
    float*  sg    = fw + F_SG;
    float*  fsg   = fw + F_FSG;
    float*  hm    = fw + F_HM;
    float*  uz    = fw + F_UZ;
    float*  zc    = fw + F_ZC;
    float*  siluz = fw + F_SILUZ;
    float*  us    = fw + F_US;
    float*  dbc   = fw + F_DBC;
    float*  dtb   = fw + F_DTB;
    float*  yf    = fw + F_YF;
    float*  yr    = fw + F_YR;
    float*  outb  = fw + F_OUTB;
    float*  proj  = fw + F_PROJ;
    bf16_t* bb    = (bf16_t*)(fw + F_END);
    bf16_t* hmbf  = bb + B_HM;
    bf16_t* y2bf  = bb + B_Y2;
    bf16_t* usbf  = bb + B_US;
    bf16_t* dbcbf = bb + B_DBC;
    bf16_t* aobf  = bb + B_AOUT;

    // ---- setup ----
    k_make_order<<<1, 1, 0, stream>>>(ordp, invp);
    k_patch_embed<<<CDIV(NB * LL * DD, 256), 256, 0, stream>>>(x, patch_w, patch_b, h);
    k_time_base<<<CDIV(NB * 256, 256), 256, 0, stream>>>(t, tin);
    k_small_gemm<<<CDIV(NB * DD, 256), 256, 0, stream>>>(tin, t_w1, t_b1, t1, NB, DD, 256, 1);
    k_small_gemm<<<CDIV(NB * DD, 256), 256, 0, stream>>>(t1, t_w2, t_b2, temb, NB, DD, DD, 0);
    k_build_cact<<<CDIV(NB * 2 * DD, 256), 256, 0, stream>>>(temb, y, cact);
    // y2 -> bf16 once (A of cond GEMM, identical for all layers)
    k_cvt_bf16<<<CDIV(MROWS * DD / 4, 256), 256, 0, stream>>>(y2, y2bf, MROWS * DD / 4);

    // ---- one-time weight transpose+convert to bf16 (stay L2-resident) ----
    for (int layer = 0; layer < DEPTH; ++layer) {
        k_tconv_bf16<<<CDIV(DD * 2 * EE, 256), 256, 0, stream>>>(
            b_in_w + (size_t)layer * DD * 2 * EE, bb + B_WTIN + (size_t)layer * 2 * EE * DD, DD, 2 * EE);
        k_tconv_bf16<<<CDIV(DD * EE, 256), 256, 0, stream>>>(
            b_cond_w + (size_t)layer * DD * EE, bb + B_WTCO + (size_t)layer * EE * DD, DD, EE);
        k_tconv_bf16<<<CDIV(EE * (RR + 2 * SS), 256), 256, 0, stream>>>(
            b_xproj_w + (size_t)layer * EE * (RR + 2 * SS),
            bb + B_WTXP + (size_t)layer * (RR + 2 * SS) * EE, EE, RR + 2 * SS);
        k_tconv_bf16<<<CDIV(RR * EE, 256), 256, 0, stream>>>(
            b_dt_w + (size_t)layer * RR * EE, bb + B_WTDT + (size_t)layer * EE * RR, RR, EE);
        k_tconv_bf16<<<CDIV(EE * DD, 256), 256, 0, stream>>>(
            b_out_w + (size_t)layer * EE * DD, bb + B_WTOU + (size_t)layer * DD * EE, EE, DD);
    }

    // ---- Mamba blocks ----
    for (int layer = 0; layer < DEPTH; ++layer) {
        const float* ada_w  = b_ada_w  + (size_t)layer * 2 * DD * 3 * DD;
        const float* ada_b  = b_ada_b  + (size_t)layer * 3 * DD;
        const float* in_b   = b_in_b   + (size_t)layer * 2 * EE;
        const float* cond_b = b_cond_b + (size_t)layer * EE;
        const float* dt_b   = b_dt_b   + (size_t)layer * EE;
        const float* Alog   = b_Alog   + (size_t)layer * EE * SS;
        const float* Dsk    = b_D      + (size_t)layer * EE;
        const float* out_b  = b_out_b  + (size_t)layer * DD;
        const bf16_t* wtin  = bb + B_WTIN + (size_t)layer * 2 * EE * DD;
        const bf16_t* wtco  = bb + B_WTCO + (size_t)layer * EE * DD;
        const bf16_t* wtxp  = bb + B_WTXP + (size_t)layer * (RR + 2 * SS) * EE;
        const bf16_t* wtdt  = bb + B_WTDT + (size_t)layer * EE * RR;
        const bf16_t* wtou  = bb + B_WTOU + (size_t)layer * DD * EE;

        // adaLN params: [shift | scale | gate] per batch
        k_small_gemm<<<CDIV(NB * 3 * DD, 256), 256, 0, stream>>>(cact, ada_w, ada_b, sg,
                                                                 NB, 3 * DD, 2 * DD, 0);
        k_ln_mod<<<MROWS, 128, 0, stream>>>(h, sg, 3 * DD, 0, DD, nullptr, hmbf);
        // uz = hm @ in_w + in_b   (1568x512 * 512x2048)
        k_wmma_gemm32x64<<<(MROWS / 32) * (2 * EE / 64), 32, 0, stream>>>(
            hmbf, wtin, in_b, uz, MROWS, 2 * EE, DD, DD, DD, 2 * EE, 0);
        // zc = y2 @ cond_w + cond_b   (1568x512 * 512x1024)
        k_wmma_gemm32x64<<<(MROWS / 32) * (EE / 64), 32, 0, stream>>>(
            y2bf, wtco, cond_b, zc, MROWS, EE, DD, DD, DD, EE, 0);
        k_zsilu<<<CDIV(MROWS * EE, 256), 256, 0, stream>>>(uz, zc, w, siluz);
        k_gather<<<CDIV(MROWS * EE, 256), 256, 0, stream>>>(uz, ordp, us, usbf);
        // dbc = u_s @ xproj   (1568x1024 * 1024x64)
        k_wmma_gemm32x64<<<(MROWS / 32) * ((RR + 2 * SS) / 64), 32, 0, stream>>>(
            usbf, wtxp, nullptr, dbc, MROWS, RR + 2 * SS, EE, EE, EE, RR + 2 * SS, 0);
        k_cvt_bf16<<<CDIV(MROWS * (RR + 2 * SS) / 4, 256), 256, 0, stream>>>(
            dbc, dbcbf, MROWS * (RR + 2 * SS) / 4);
        // dt = softplus(dbc[:, :32] @ dt_w + dt_b)   (K=32 -> single WMMA step)
        k_wmma_gemm32x64<<<(MROWS / 32) * (EE / 64), 32, 0, stream>>>(
            dbcbf, wtdt, dt_b, dtb, MROWS, EE, RR, RR + 2 * SS, RR, EE, 1);
        // bidirectional selective scan
        k_scan<<<CDIV(NB * EE, 256), 256, 0, stream>>>(us, dtb, dbc, Alog, yf, 0);
        k_scan<<<CDIV(NB * EE, 256), 256, 0, stream>>>(us, dtb, dbc, Alog, yr, 1);
        k_combine<<<CDIV(MROWS * EE, 256), 256, 0, stream>>>(yf, yr, us, Dsk, siluz, invp, aobf);
        // out = aout @ out_w + out_b   (1568x1024 * 1024x512)
        k_wmma_gemm32x64<<<(MROWS / 32) * (DD / 64), 32, 0, stream>>>(
            aobf, wtou, out_b, outb, MROWS, DD, EE, EE, EE, DD, 0);
        k_residual<<<CDIV(MROWS * DD, 256), 256, 0, stream>>>(h, sg, outb);
    }

    // ---- final head ----
    k_small_gemm<<<CDIV(NB * 2 * DD, 256), 256, 0, stream>>>(cact, f_ada_w, f_ada_b, fsg,
                                                             NB, 2 * DD, 2 * DD, 0);
    k_ln_mod<<<MROWS, 128, 0, stream>>>(h, fsg, 2 * DD, 0, DD, hm, nullptr);
    k_wmma_gemm16<<<(MROWS / 16) * (PP * PP * COUT / 16), 32, 0, stream>>>(
        hm, f_lin_w, f_lin_b, proj, MROWS, PP * PP * COUT, DD, DD,
        PP * PP * COUT, PP * PP * COUT, 0);
    k_unpatchify<<<CDIV(NB * COUT * IMG * IMG, 256), 256, 0, stream>>>(proj, (float*)d_out);
}
